// AffineSTN3D_11544872091836
// MI455X (gfx1250) — matrix operand order repaired and checked by reading, not compiled, not run on hardware
//
#include <hip/hip_runtime.h>

typedef __attribute__((ext_vector_type(16))) __bf16 v16bf;
typedef __attribute__((ext_vector_type(8)))  __bf16 v8bf;
typedef __attribute__((ext_vector_type(8)))  float  v8f;
typedef __attribute__((ext_vector_type(4)))  unsigned int u32x4;
typedef __attribute__((ext_vector_type(8)))  int i32x8;
typedef __attribute__((ext_vector_type(4)))  int i32x4;

#define PI4_F 0.78539816339744830962f

// ---------------------------------------------------------------- utilities
__global__ void zero_f32(float* __restrict__ p, int n) {
  int i = blockIdx.x * blockDim.x + threadIdx.x;
  if (i < n) p[i] = 0.f;
}

__global__ void f32_to_bf16(const float* __restrict__ in, __bf16* __restrict__ out, int n) {
  int i = blockIdx.x * blockDim.x + threadIdx.x;
  if (i < n) out[i] = (__bf16)in[i];
}

// Repack conv weights (Cout, Cin, 27) fp32 -> bf16 [Cout][Kpad], K = tap*Cin+ci.
__global__ void repack_w(const float* __restrict__ w, __bf16* __restrict__ wp,
                         int Cin, int Cout, int Kreal, int Kpad) {
  int i = blockIdx.x * blockDim.x + threadIdx.x;
  if (i >= Kpad * Cout) return;
  int co = i / Kpad, k = i - co * Kpad;
  __bf16 v = (__bf16)0.f;
  if (k < Kreal) {
    int tap = k / Cin, ci = k - tap * Cin;
    v = (__bf16)w[(co * Cin + ci) * 27 + tap];
  }
  wp[i] = v;
}

// ------------------------------------------------- implicit-GEMM conv (WMMA)
// Channels-last activations (NDHWC). M = B*Do*Ho*Wo, N = Cout, K = 27*Cin
// (tap-major, padded to 32). Weights [Cout][Kpad] staged into LDS by the
// Tensor Data Mover once per block. Each wave register-blocks MREP 16-row
// tiles sharing one B fragment -> MREP independent WMMAs per K chunk.
// All A-side predication is branchless (cndmask'd offsets + value selects).
template <int MREP, bool BIGC>
__global__ __launch_bounds__(128) void conv3d_wmma_bf16(
    const __bf16* __restrict__ in, const __bf16* __restrict__ wp,
    const float* __restrict__ bias, __bf16* __restrict__ out,
    int Bn, int Cin, int cinShift, int Di, int Hi, int Wi,
    int Cout, int Do_, int Ho, int Wo, int stride, int Kreal, int Kpad)
{
  extern __shared__ __bf16 ldsW[];           // [16 columns][Kpad] (n-major)
  const int lane = threadIdx.x & 31;
  const int wave = threadIdx.x >> 5;
  const int half = lane >> 4;                // 0: lanes 0-15, 1: lanes 16-31
  const int l16  = lane & 15;
  const int ntile = blockIdx.y;

  // ---- TDM: DMA this block's 16-output-channel weight slice into LDS.
  if (threadIdx.x < 32) {
    unsigned long long ga = (unsigned long long)(const void*)wp +
                            (unsigned long long)((unsigned)ntile * 16u * (unsigned)Kpad * 2u);
    u32x4 g0; i32x8 g1; i32x4 g2, g3; i32x8 g4;
    for (int q = 0; q < 4; ++q) { g2[q] = 0; g3[q] = 0; }
    for (int q = 0; q < 8; ++q) g4[q] = 0;
    const unsigned td0 = (unsigned)Kpad;     // tensor dim0 (elements, contiguous K)
    const unsigned td1 = 16u;                // tensor dim1 (output channels)
    const unsigned tile0 = (unsigned)Kpad;   // tile dim0
    const unsigned tile1 = 16u;              // tile dim1
    g0[0] = 1u;                                            // count=1 (valid D#)
    g0[1] = 0u;                                            // lds_addr = 0 (dyn LDS base)
    g0[2] = (unsigned)(ga & 0xFFFFFFFFull);                // global_addr lo
    g0[3] = (unsigned)((ga >> 32) & 0x01FFFFFFull) | 0x80000000u; // addr hi | type=2
    g1[0] = (int)(1u << 16);                               // data_size = 2 bytes
    g1[1] = (int)((td0 & 0xFFFFu) << 16);                  // tensor_dim0[15:0]
    g1[2] = (int)((td0 >> 16) | ((td1 & 0xFFFFu) << 16));  // td0 hi | td1 lo
    g1[3] = (int)((td1 >> 16) | (tile0 << 16));            // td1 hi | tile_dim0
    g1[4] = (int)tile1;                                    // tile_dim1 | tile_dim2=0
    g1[5] = (int)td0;                                      // dim0 stride lo (= Kpad)
    g1[6] = 0;                                             // stride hi | dim1 stride=0
    g1[7] = 0;
    __builtin_amdgcn_tensor_load_to_lds(g0, g1, g2, g3, g4, 0);
    __builtin_amdgcn_s_wait_tensorcnt((short)0);
  }
  __syncthreads();

  const int M = Bn * Do_ * Ho * Wo;
  const int mt0 = (blockIdx.x * 4 + wave) * MREP;
  if (mt0 * 16 >= M) return;
  const int nn = ntile * 16 + l16;           // B/C/D column for this lane

  int bb[MREP], id0[MREP], ih0[MREP], iw0[MREP];
#pragma unroll
  for (int r = 0; r < MREP; ++r) {
    int t = (mt0 + r) * 16 + l16;            // A row for this lane, tile r
    int ow = t % Wo; t /= Wo;
    int oh = t % Ho; t /= Ho;
    int od = t % Do_; bb[r] = t / Do_;
    id0[r] = od * stride - 1;
    ih0[r] = oh * stride - 1;
    iw0[r] = ow * stride - 1;
  }

  const __bf16* ldsB = ldsW + l16 * Kpad;
  const int cmask = Cin - 1;
  const int ksteps = Kpad >> 5;
  v8f c[MREP];
#pragma unroll
  for (int r = 0; r < MREP; ++r) {
    v8f z = {};
    c[r] = z;
  }

  for (int kk = 0; kk < ksteps; ++kk) {
    const int kbase = kk << 5;
    v16bf a[MREP];
    if (BIGC) {
      // ---- A fragments: two contiguous 8-element (16B) runs per lane/tile.
#pragma unroll
      for (int rrun = 0; rrun < 2; ++rrun) {
        const int kR  = kbase + 16 * rrun;
        const int tap = kR >> cinShift;               // wave-uniform
        const int ci  = (kR & cmask) + half * 8;
        const int tok = (int)(tap < 27);
        const int tapc = tok ? tap : 0;
        const int kd = tapc / 9; const int rr2 = tapc - kd * 9;
        const int kh = rr2 / 3, kw = rr2 - kh * 3;
#pragma unroll
        for (int r = 0; r < MREP; ++r) {
          const int id = id0[r] + kd, ih = ih0[r] + kh, iw = iw0[r] + kw;
          const int ok = tok & (int)((unsigned)id < (unsigned)Di) &
                         (int)((unsigned)ih < (unsigned)Hi) &
                         (int)((unsigned)iw < (unsigned)Wi);
          const int offv = (((bb[r] * Di + id) * Hi + ih) * Wi + iw) * Cin + ci;
          const int off = ok ? offv : 0;               // safe address, no branch
          v8bf v = *(const v8bf*)(in + off);
#pragma unroll
          for (int q = 0; q < 8; ++q)
            a[r][rrun * 8 + q] = ok ? v[q] : (__bf16)0.f;
        }
      }
    } else {
      // Cin == 1 (first layer): single K chunk; per-element gather with the
      // tap decomposition folded to compile-time constants selected by `half`.
#pragma unroll
      for (int r = 0; r < MREP; ++r) {
#pragma unroll
        for (int j = 0; j < 16; ++j) {
          const int b0 = (j < 8) ? j : (8 + j);        // constexpr after unroll
          const int kA = b0, kB = b0 + 8;
          const int rA = kA - (kA / 9) * 9, rB = kB - (kB / 9) * 9;
          const int k  = half ? kB : kA;
          const int kd = half ? (kB / 9) : (kA / 9);
          const int kh = half ? (rB / 3) : (rA / 3);
          const int kw = half ? (rB - (rB / 3) * 3) : (rA - (rA / 3) * 3);
          const int id = id0[r] + kd, ih = ih0[r] + kh, iw = iw0[r] + kw;
          const int ok = (int)(k < Kreal) & (int)((unsigned)id < (unsigned)Di) &
                         (int)((unsigned)ih < (unsigned)Hi) &
                         (int)((unsigned)iw < (unsigned)Wi);
          const int offv = ((bb[r] * Di + id) * Hi + ih) * Wi + iw;
          const int off = ok ? offv : 0;               // safe address, no branch
          __bf16 v = in[off];
          a[r][j] = ok ? v : (__bf16)0.f;
        }
      }
    }
    // ---- shared B fragment: two 16B runs from LDS (k = 8h+s, 16+8h+s).
    v16bf bm;
    const __bf16* bp = ldsB + kbase + half * 8;
#pragma unroll
    for (int q = 0; q < 16; ++q) bm[q] = bp[(q < 8) ? q : (q + 8)];
    // ---- MREP independent WMMAs back-to-back.
#pragma unroll
    for (int r = 0; r < MREP; ++r)
      c[r] = __builtin_amdgcn_wmma_f32_16x16x32_bf16(false, a[r], false, bm,
                                                     (short)0, c[r], false, false);
  }

  // ---- bias + store, channels-last: out[m*Cout + n]; coalesced across lanes.
  const float bv = bias[nn];
#pragma unroll
  for (int r = 0; r < MREP; ++r) {
#pragma unroll
    for (int i = 0; i < 8; ++i) {
      int mr = (mt0 + r) * 16 + i + half * 8;  // lanes 16-31 hold rows M=i+8
      out[mr * Cout + nn] = (__bf16)(c[r][i] + bv);
    }
  }
}

// ---------------------------------------------- per-channel BN stats + apply
__global__ __launch_bounds__(256) void bn_stats(const __bf16* __restrict__ act,
                                                float* __restrict__ stats, // [2C]
                                                int C, int cshift, int rows) {
  const int c = threadIdx.x & (C - 1);
  const int g = threadIdx.x >> cshift;
  const int R = 256 >> cshift;
  float s1 = 0.f, s2 = 0.f;
  for (int r = blockIdx.x * R + g; r < rows; r += gridDim.x * R) {
    float v = (float)act[r * C + c];
    s1 += v; s2 += v * v;
  }
  __shared__ float m1[256], m2[256];
  m1[threadIdx.x] = s1; m2[threadIdx.x] = s2;
  __syncthreads();
  if (g == 0) {
    for (int q = 1; q < R; ++q) {
      s1 += m1[(q << cshift) + c];
      s2 += m2[(q << cshift) + c];
    }
    atomicAdd(&stats[c], s1);
    atomicAdd(&stats[C + c], s2);
  }
}

__global__ void bn_relu_kernel(__bf16* __restrict__ act, const float* __restrict__ stats,
                               const float* __restrict__ g, const float* __restrict__ be,
                               int C, int total, float invCnt) {
  int i = blockIdx.x * blockDim.x + threadIdx.x;
  if (i >= total) return;
  int cch = i & (C - 1);
  float mean = stats[cch] * invCnt;
  float var  = stats[C + cch] * invCnt - mean * mean;
  float x = (float)act[i];
  float y = (x - mean) * rsqrtf(var + 1e-5f) * g[cch] + be[cch];
  act[i] = (__bf16)fmaxf(y, 0.f);
}

// --------------------------------------------------- affine head (per batch)
__device__ inline void mm4(const float* A, const float* B, float* C) {
  for (int i = 0; i < 4; ++i)
    for (int j = 0; j < 4; ++j) {
      float s = 0.f;
      for (int k = 0; k < 4; ++k) s += A[i * 4 + k] * B[k * 4 + j];
      C[i * 4 + j] = s;
    }
}
__device__ inline void eye4(float* A) {
  for (int i = 0; i < 16; ++i) A[i] = (i % 5 == 0) ? 1.f : 0.f;
}

// xs is channels-last [B][64 spatial][64 ch]; NCDHW flatten index f = c*64+s
// maps to xb[(f&63)*64 + (f>>6)].
__global__ __launch_bounds__(256) void fc_affine_kernel(
    const __bf16* __restrict__ xs,
    const float* __restrict__ tw, const float* __restrict__ tb,
    const float* __restrict__ rw, const float* __restrict__ rb,
    const float* __restrict__ sw, const float* __restrict__ sb,
    const float* __restrict__ shw, const float* __restrict__ shb,
    float* __restrict__ theta, int F)
{
  const int b = blockIdx.x;
  const int tid = threadIdx.x;
  float acc[12];
#pragma unroll
  for (int o = 0; o < 12; ++o) acc[o] = 0.f;
  const __bf16* xb = xs + b * F;
  const float* Ws[4] = {tw, rw, sw, shw};
  for (int i = tid; i < F; i += 256) {
    float xv = (float)xb[((i & 63) << 6) + (i >> 6)];
#pragma unroll
    for (int h = 0; h < 4; ++h) {
      const float* W = Ws[h];
      acc[h * 3 + 0] += xv * W[0 * F + i];
      acc[h * 3 + 1] += xv * W[1 * F + i];
      acc[h * 3 + 2] += xv * W[2 * F + i];
    }
  }
  __shared__ float red[12][256];
  for (int o = 0; o < 12; ++o) red[o][tid] = acc[o];
  __syncthreads();
  for (int s = 128; s > 0; s >>= 1) {
    if (tid < s)
      for (int o = 0; o < 12; ++o) red[o][tid] += red[o][tid + s];
    __syncthreads();
  }
  if (tid != 0) return;

  float trans[3], rot[3], scale[3], sh[3];
  for (int j = 0; j < 3; ++j) {
    trans[j] = tanhf(red[0 + j][0] + tb[j])  * 0.1f;
    rot[j]   = tanhf(red[3 + j][0] + rb[j])  * PI4_F;
    scale[j] = tanhf(red[6 + j][0] + sb[j])  * 0.2f;
    sh[j]    = tanhf(red[9 + j][0] + shb[j]) * PI4_F;
  }
  float T[16]; eye4(T);
  T[3] = trans[0]; T[7] = trans[1]; T[11] = trans[2];

  float R1[16], R2[16], R3[16], R[16], tmp[16];
  eye4(R1); R1[0] = cosf(rot[0]); R1[1] = -sinf(rot[0]); R1[4] = sinf(rot[0]); R1[5] = cosf(rot[0]);
  eye4(R2); R2[5] = cosf(rot[1]); R2[6] = -sinf(rot[1]); R2[9] = sinf(rot[1]); R2[10] = cosf(rot[1]);
  eye4(R3); R3[0] = cosf(rot[2]); R3[1] = -sinf(rot[2]); R3[4] = sinf(rot[2]); R3[5] = cosf(rot[2]);
  mm4(R1, R2, tmp); mm4(tmp, R3, R);

  float S[16]; eye4(S);
  S[0] = expf(scale[0]); S[5] = expf(scale[1]); S[10] = expf(scale[2]);

  float H1[16], H2[16], H3[16], Sh[16];
  eye4(H1); H1[5] = cosf(sh[0]); H1[6] = -sinf(sh[0]); H1[9] = sinf(sh[0]); H1[10] = cosf(sh[0]);
  eye4(H2); H2[0] = cosf(sh[1]); H2[2] =  sinf(sh[1]); H2[8] = -sinf(sh[1]); H2[10] = cosf(sh[1]);
  eye4(H3); H3[0] = cosf(sh[2]); H3[1] = -sinf(sh[2]); H3[4] = sinf(sh[2]); H3[5] = cosf(sh[2]);
  mm4(H1, H2, tmp); mm4(tmp, H3, Sh);

  float ShT[16];
  for (int i = 0; i < 4; ++i)
    for (int j = 0; j < 4; ++j) ShT[i * 4 + j] = Sh[j * 4 + i];

  float A1[16], A2[16], A3[16], Mf[16];
  mm4(Sh, S, A1); mm4(A1, ShT, A2); mm4(A2, R, A3); mm4(A3, T, Mf);
  for (int r = 0; r < 3; ++r)
    for (int cidx = 0; cidx < 4; ++cidx)
      theta[b * 12 + r * 4 + cidx] = Mf[r * 4 + cidx];
}

// ---------------------------------------------------------------- resampling
__global__ void grid_sample_kernel(const float* __restrict__ img,
                                   const float* __restrict__ theta,
                                   float* __restrict__ out,
                                   int Bn, int D, int H, int W) {
  int idx = blockIdx.x * blockDim.x + threadIdx.x;
  int vox = D * H * W;
  if (idx >= Bn * vox) return;
  int t = idx % vox;
  int b = idx / vox;
  int w = t % W; t /= W;
  int h = t % H; t /= H;
  int d = t;
  float xn = -1.f + 2.f * w / (float)(W - 1);
  float yn = -1.f + 2.f * h / (float)(H - 1);
  float zn = -1.f + 2.f * d / (float)(D - 1);
  const float* th = theta + b * 12;
  float gx = th[0] * xn + th[1] * yn + th[2]  * zn + th[3];
  float gy = th[4] * xn + th[5] * yn + th[6]  * zn + th[7];
  float gz = th[8] * xn + th[9] * yn + th[10] * zn + th[11];
  float fx = ((gx + 1.f) * W - 1.f) * 0.5f;
  float fy = ((gy + 1.f) * H - 1.f) * 0.5f;
  float fz = ((gz + 1.f) * D - 1.f) * 0.5f;
  float x0 = floorf(fx), y0 = floorf(fy), z0 = floorf(fz);
  const float* im = img + b * vox;
  float o = 0.f;
  for (int dz = 0; dz < 2; ++dz) {
    float zc = z0 + dz;
    float wz = dz ? (fz - z0) : (z0 + 1.f - fz);
    bool zv = (zc >= 0.f) && (zc <= (float)(D - 1));
    int zi = (int)fminf(fmaxf(zc, 0.f), (float)(D - 1));
    for (int dy = 0; dy < 2; ++dy) {
      float yc = y0 + dy;
      float wy = dy ? (fy - y0) : (y0 + 1.f - fy);
      bool yv = (yc >= 0.f) && (yc <= (float)(H - 1));
      int yi = (int)fminf(fmaxf(yc, 0.f), (float)(H - 1));
      for (int dx = 0; dx < 2; ++dx) {
        float xc = x0 + dx;
        float wx = dx ? (fx - x0) : (x0 + 1.f - fx);
        bool xv = (xc >= 0.f) && (xc <= (float)(W - 1));
        int xi = (int)fminf(fmaxf(xc, 0.f), (float)(W - 1));
        float wgt = (zv && yv && xv) ? (wz * wy * wx) : 0.f;
        o += wgt * im[(zi * H + yi) * W + xi];
      }
    }
  }
  out[idx] = o;
}

// ------------------------------------------------------------------- launch
extern "C" void kernel_launch(void* const* d_in, const int* in_sizes, int n_in,
                              void* d_out, int out_size, void* d_ws, size_t ws_size,
                              hipStream_t stream) {
  const float* x = (const float*)d_in[0];
  const float* w[5]  = {(const float*)d_in[1],  (const float*)d_in[5],
                        (const float*)d_in[9],  (const float*)d_in[13],
                        (const float*)d_in[17]};
  const float* bi[5] = {(const float*)d_in[2],  (const float*)d_in[6],
                        (const float*)d_in[10], (const float*)d_in[14],
                        (const float*)d_in[18]};
  const float* g[5]  = {(const float*)d_in[3],  (const float*)d_in[7],
                        (const float*)d_in[11], (const float*)d_in[15],
                        (const float*)d_in[19]};
  const float* be[5] = {(const float*)d_in[4],  (const float*)d_in[8],
                        (const float*)d_in[12], (const float*)d_in[16],
                        (const float*)d_in[20]};
  const float* tw  = (const float*)d_in[21];
  const float* tb  = (const float*)d_in[22];
  const float* rw  = (const float*)d_in[23];
  const float* rb  = (const float*)d_in[24];
  const float* sw  = (const float*)d_in[25];
  const float* sb  = (const float*)d_in[26];
  const float* shw = (const float*)d_in[27];
  const float* shb = (const float*)d_in[28];

  const int Bn = 16, S = 64;
  struct L { int Cin, cinShift, Cout, coutShift, Di, Do, stride, Kreal, Kpad; };
  const L ls[5] = {
    { 1, 0, 16, 4, 64, 64, 1,   27,   32},
    {16, 4, 32, 5, 64, 32, 2,  432,  448},
    {32, 5, 64, 6, 32, 16, 2,  864,  864},
    {64, 6, 64, 6, 16,  8, 2, 1728, 1728},
    {64, 6, 64, 6,  8,  4, 2, 1728, 1728},
  };

  char* ws = (char*)d_ws;
  size_t off = 0;
  auto alloc = [&](size_t bytes) -> char* {
    char* p = ws + off;
    off = (off + bytes + 255) & ~(size_t)255;
    return p;
  };

  __bf16* xbf = (__bf16*)alloc((size_t)Bn * S * S * S * 2);
  __bf16* act[5];
  for (int i = 0; i < 5; ++i) {
    size_t Do = ls[i].Do;
    act[i] = (__bf16*)alloc((size_t)Bn * ls[i].Cout * Do * Do * Do * 2);
  }
  __bf16* wp[5];
  for (int i = 0; i < 5; ++i)
    wp[i] = (__bf16*)alloc((size_t)ls[i].Kpad * ls[i].Cout * 2);
  const int nstats = 2 * (16 + 32 + 64 + 64 + 64);
  float* statsBase = (float*)alloc((size_t)nstats * 4);
  float* stats[5];
  {
    float* p = statsBase;
    for (int i = 0; i < 5; ++i) { stats[i] = p; p += 2 * ls[i].Cout; }
  }
  float* theta = (float*)alloc((size_t)Bn * 12 * 4);

  zero_f32<<<(nstats + 255) / 256, 256, 0, stream>>>(statsBase, nstats);

  int nx = Bn * S * S * S;
  f32_to_bf16<<<(nx + 255) / 256, 256, 0, stream>>>(x, xbf, nx);

  for (int i = 0; i < 5; ++i) {
    int total = ls[i].Kpad * ls[i].Cout;
    repack_w<<<(total + 255) / 256, 256, 0, stream>>>(w[i], wp[i], ls[i].Cin,
                                                      ls[i].Cout, ls[i].Kreal,
                                                      ls[i].Kpad);
  }

  const __bf16* cur = xbf;
  int Di = S;
  for (int i = 0; i < 5; ++i) {
    int Do = ls[i].Do;
    int M = Bn * Do * Do * Do;
    int mrep = (i == 0) ? 1 : 4;
    int tilesPerBlock = 4 * mrep;             // 4 waves x MREP tiles
    dim3 grid((unsigned)((M / 16 + tilesPerBlock - 1) / tilesPerBlock),
              (unsigned)(ls[i].Cout / 16));
    size_t ldsBytes = (size_t)ls[i].Kpad * 16 * 2;   // weight tile in LDS
    if (i == 0) {
      conv3d_wmma_bf16<1, false><<<grid, 128, ldsBytes, stream>>>(
          cur, wp[i], bi[i], act[i],
          Bn, ls[i].Cin, ls[i].cinShift, Di, Di, Di,
          ls[i].Cout, Do, Do, Do, ls[i].stride, ls[i].Kreal, ls[i].Kpad);
    } else {
      conv3d_wmma_bf16<4, true><<<grid, 128, ldsBytes, stream>>>(
          cur, wp[i], bi[i], act[i],
          Bn, ls[i].Cin, ls[i].cinShift, Di, Di, Di,
          ls[i].Cout, Do, Do, Do, ls[i].stride, ls[i].Kreal, ls[i].Kpad);
    }

    int spatial = Do * Do * Do;
    int rows = Bn * spatial;
    bn_stats<<<256, 256, 0, stream>>>(act[i], stats[i], ls[i].Cout,
                                      ls[i].coutShift, rows);

    int total = rows * ls[i].Cout;
    float invCnt = 1.f / (float)rows;
    bn_relu_kernel<<<(total + 255) / 256, 256, 0, stream>>>(
        act[i], stats[i], g[i], be[i], ls[i].Cout, total, invCnt);
    cur = act[i];
    Di = Do;
  }

  fc_affine_kernel<<<Bn, 256, 0, stream>>>(act[4], tw, tb, rw, rb, sw, sb,
                                           shw, shb, theta, 4096);

  int nout = Bn * S * S * S;
  grid_sample_kernel<<<(nout + 255) / 256, 256, 0, stream>>>(
      x, theta, (float*)d_out, Bn, S, S, S);
}
